// MultiheadGEAttention_80496277062282
// MI455X (gfx1250) — matrix operand order repaired and checked by reading, not compiled, not run on hardware
//
#include <hip/hip_runtime.h>
#include <hip/hip_bf16.h>

// ---------------------------------------------------------------------------
// MultiheadGEAttention for MI455X (gfx1250, wave32, WMMA)
// B=32 H=8 NQ=NKV=512 D=1024 DH=128 N_COLS=512 D_COL=18
// ---------------------------------------------------------------------------

typedef __attribute__((ext_vector_type(16))) _Float16 v16h;
typedef __attribute__((ext_vector_type(8)))  _Float16 v8h;
typedef __attribute__((ext_vector_type(8)))  float    v8f;
typedef __attribute__((ext_vector_type(4)))  float    v4f;

#define WMMA_F32_F16(a, b, c) \
  __builtin_amdgcn_wmma_f32_16x16x32_f16(false, (a), false, (b), (short)0, (c), false, false)

#define CB   32
#define CH   8
#define CN   512     // NQ == NKV == N_COLS
#define CD   1024
#define CDH  128
#define CDCOL 18

// LDS row pitch in halves: 72 halves = 144 B.
//  - rows are 16B-aligned -> b128 LDS stores
//  - 36 dwords/row: m*36 mod 64 covers 16 distinct banks for 16-lane row access
#define LDSW 72
#define KT   64      // K-tile (two 16x16x32 WMMA k-steps per barrier)

// --- helpers ---------------------------------------------------------------
__device__ __forceinline__ v8h cvt8(v4f f0, v4f f1) {
  v8h h;
#pragma unroll
  for (int j = 0; j < 4; ++j) { h[j] = (_Float16)f0[j]; h[4 + j] = (_Float16)f1[j]; }
  return h;
}

// --- WMMA fragment loaders (wave32; A and B share contiguous k = 16g + s) ---------------------
// A 16x32 f16: lane m = lane%16, g = lane/16; VGPR v<4 -> K=2v+16g, v>=4 -> K=8+2(v-4)+16g
__device__ __forceinline__ v16h frag_a_row(const _Float16* row, int g) {
  v16h a;
#pragma unroll
  for (int v = 0; v < 4; ++v) {
    int kb = 2 * v + 16 * g;
    a[2 * v]     = row[kb];
    a[2 * v + 1] = row[kb + 1];
    int kb2 = 8 + 2 * v + 16 * g;
    a[8 + 2 * v]     = row[kb2];
    a[8 + 2 * v + 1] = row[kb2 + 1];
  }
  return a;
}
// B 32x16 f16 (column-major per lane): lane n = lane%16 holds column n; K = 2v + 16g
__device__ __forceinline__ v16h frag_b_row(const _Float16* row, int g) {
  v16h b;
#pragma unroll
  for (int v = 0; v < 8; ++v) {
    int kb = 2 * v + 16 * g;
    b[2 * v]     = row[kb];
    b[2 * v + 1] = row[kb + 1];
  }
  return b;
}
// B fragment from a [kk][n] row-major LDS tile (element (kk,n) at base[kk*LDSW+n])
__device__ __forceinline__ v16h frag_b_col(const _Float16* base, int n, int g) {
  v16h b;
#pragma unroll
  for (int v = 0; v < 8; ++v) {
    int kb = 2 * v + 16 * g;
    b[2 * v]     = base[kb * LDSW + n];
    b[2 * v + 1] = base[(kb + 1) * LDSW + n];
  }
  return b;
}

// ---------------------------------------------------------------------------
// Generic projection GEMM: Y[m,n] = (sum_k A[m,k] * W[n,k] + bias[n]) * scale[n]
// Block: 256 threads = 8 waves (4 M-waves x 2 N-waves), tile 128x64, K step 64.
// Register double-buffer: tile k+1 is prefetched into VGPRs during compute of k.
// ---------------------------------------------------------------------------
template <int A_F16, int OUT_F16>
__global__ __launch_bounds__(256) void proj_gemm(
    const void* __restrict__ Ain,
    const float* __restrict__ W, const float* __restrict__ bias,
    const float* __restrict__ scale,
    void* __restrict__ Out,
    int M, int N, int K)
{
  __shared__ _Float16 As[128 * LDSW];
  __shared__ _Float16 Bs[64 * LDSW];
  const int tid  = threadIdx.x;
  const int m0   = blockIdx.y * 128;
  const int n0   = blockIdx.x * 64;
  const int wave = tid >> 5, lane = tid & 31;
  const int wm = wave & 3, wn = wave >> 2;
  const int g  = lane >> 4, l16 = lane & 15;

  const float*    A32 = (const float*)Ain;
  const _Float16* A16 = (const _Float16*)Ain;

  v8h a_reg[4];   // A tile: 128x64 halves = 1024 segs of 8 -> 4/thread
  v8h b_reg[2];   // B tile:  64x64 halves =  512 segs of 8 -> 2/thread

  auto loadA = [&](int k0) {
#pragma unroll
    for (int p = 0; p < 4; ++p) {
      int idx = tid + p * 256;
      int m = idx >> 3, kk = (idx & 7) * 8;
      if (A_F16) {
        a_reg[p] = *(const v8h*)(A16 + (size_t)(m0 + m) * K + k0 + kk);
      } else {
        const float* src = A32 + (size_t)(m0 + m) * K + k0 + kk;
        a_reg[p] = cvt8(*(const v4f*)src, *(const v4f*)(src + 4));
      }
    }
  };
  auto loadB = [&](int k0) {
#pragma unroll
    for (int p = 0; p < 2; ++p) {
      int idx = tid + p * 256;
      int n = idx >> 3, kk = (idx & 7) * 8;
      const float* src = W + (size_t)(n0 + n) * K + k0 + kk;
      b_reg[p] = cvt8(*(const v4f*)src, *(const v4f*)(src + 4));
    }
  };
  auto storeLDS = [&]() {
#pragma unroll
    for (int p = 0; p < 4; ++p) {
      int idx = tid + p * 256;
      int m = idx >> 3, kk = (idx & 7) * 8;
      *(v8h*)(&As[m * LDSW + kk]) = a_reg[p];
    }
#pragma unroll
    for (int p = 0; p < 2; ++p) {
      int idx = tid + p * 256;
      int n = idx >> 3, kk = (idx & 7) * 8;
      *(v8h*)(&Bs[n * LDSW + kk]) = b_reg[p];
    }
  };

  v8f acc[2][2] = {};
  loadA(0); loadB(0);
  const int nkt = K / KT;
  for (int kt = 0; kt < nkt; ++kt) {
    storeLDS();
    __syncthreads();
    if (kt + 1 < nkt) { loadA((kt + 1) * KT); loadB((kt + 1) * KT); }
#pragma unroll
    for (int ks = 0; ks < 2; ++ks) {
      const _Float16* ab = &As[ks * 32];
      const _Float16* bb = &Bs[ks * 32];
      v16h a0 = frag_a_row(ab + (wm * 32 + l16) * LDSW, g);
      v16h a1 = frag_a_row(ab + (wm * 32 + 16 + l16) * LDSW, g);
      v16h b0 = frag_b_row(bb + (wn * 32 + l16) * LDSW, g);
      v16h b1 = frag_b_row(bb + (wn * 32 + 16 + l16) * LDSW, g);
      acc[0][0] = WMMA_F32_F16(a0, b0, acc[0][0]);
      acc[0][1] = WMMA_F32_F16(a0, b1, acc[0][1]);
      acc[1][0] = WMMA_F32_F16(a1, b0, acc[1][0]);
      acc[1][1] = WMMA_F32_F16(a1, b1, acc[1][1]);
    }
    __syncthreads();
  }

#pragma unroll
  for (int sm = 0; sm < 2; ++sm)
#pragma unroll
    for (int sn = 0; sn < 2; ++sn) {
      int col = n0 + wn * 32 + sn * 16 + l16;
      float bv = bias  ? bias[col]  : 0.0f;
      float sv = scale ? scale[col] : 1.0f;
#pragma unroll
      for (int r = 0; r < 8; ++r) {
        int row = m0 + wm * 32 + sm * 16 + g * 8 + r;
        float y = (acc[sm][sn][r] + bv) * sv;
        if (OUT_F16) ((_Float16*)Out)[(size_t)row * N + col] = (_Float16)y;
        else         ((float*)Out)[(size_t)row * N + col]    = y;
      }
    }
}

// ---------------------------------------------------------------------------
// Score: S[bh,q,k] = (f_head . f_tail) / sqrt(DH) + adj_mask[h,q,k]
// Per (b,h): [512x128] @ [128x512]; tile 128x64, K = 128 (two K-tiles).
// ---------------------------------------------------------------------------
__global__ __launch_bounds__(256) void score_gemm(
    const _Float16* __restrict__ f_head, const _Float16* __restrict__ f_tail,
    const float* __restrict__ adjm, float* __restrict__ S)
{
  __shared__ _Float16 As[128 * LDSW];
  __shared__ _Float16 Bs[64 * LDSW];
  const int tid = threadIdx.x;
  const int q0  = blockIdx.y * 128;
  const int n0  = blockIdx.x * 64;
  const int bh  = blockIdx.z;
  const int b   = bh / CH, h = bh % CH;
  const _Float16* Abase = f_head + (size_t)b * CN * CD + h * CDH;
  const _Float16* Bbase = f_tail + (size_t)b * CN * CD + h * CDH;

  const int wave = tid >> 5, lane = tid & 31;
  const int wm = wave & 3, wn = wave >> 2;
  const int g  = lane >> 4, l16 = lane & 15;

  v8h a_reg[4], b_reg[2];
  auto loadA = [&](int k0) {
#pragma unroll
    for (int p = 0; p < 4; ++p) {
      int idx = tid + p * 256;
      int m = idx >> 3, kk = (idx & 7) * 8;
      a_reg[p] = *(const v8h*)(Abase + (size_t)(q0 + m) * CD + k0 + kk);
    }
  };
  auto loadB = [&](int k0) {
#pragma unroll
    for (int p = 0; p < 2; ++p) {
      int idx = tid + p * 256;
      int n = idx >> 3, kk = (idx & 7) * 8;
      b_reg[p] = *(const v8h*)(Bbase + (size_t)(n0 + n) * CD + k0 + kk);
    }
  };
  auto storeLDS = [&]() {
#pragma unroll
    for (int p = 0; p < 4; ++p) {
      int idx = tid + p * 256;
      int m = idx >> 3, kk = (idx & 7) * 8;
      *(v8h*)(&As[m * LDSW + kk]) = a_reg[p];
    }
#pragma unroll
    for (int p = 0; p < 2; ++p) {
      int idx = tid + p * 256;
      int n = idx >> 3, kk = (idx & 7) * 8;
      *(v8h*)(&Bs[n * LDSW + kk]) = b_reg[p];
    }
  };

  v8f acc[2][2] = {};
  loadA(0); loadB(0);
#pragma unroll
  for (int kt = 0; kt < CDH / KT; ++kt) {
    storeLDS();
    __syncthreads();
    if (kt + 1 < CDH / KT) { loadA((kt + 1) * KT); loadB((kt + 1) * KT); }
#pragma unroll
    for (int ks = 0; ks < 2; ++ks) {
      const _Float16* ab = &As[ks * 32];
      const _Float16* bb = &Bs[ks * 32];
      v16h a0 = frag_a_row(ab + (wm * 32 + l16) * LDSW, g);
      v16h a1 = frag_a_row(ab + (wm * 32 + 16 + l16) * LDSW, g);
      v16h b0 = frag_b_row(bb + (wn * 32 + l16) * LDSW, g);
      v16h b1 = frag_b_row(bb + (wn * 32 + 16 + l16) * LDSW, g);
      acc[0][0] = WMMA_F32_F16(a0, b0, acc[0][0]);
      acc[0][1] = WMMA_F32_F16(a0, b1, acc[0][1]);
      acc[1][0] = WMMA_F32_F16(a1, b0, acc[1][0]);
      acc[1][1] = WMMA_F32_F16(a1, b1, acc[1][1]);
    }
    __syncthreads();
  }

  const float inv_sqrt_dh = 0.08838834764831845f;  // 1/sqrt(128)
#pragma unroll
  for (int sm = 0; sm < 2; ++sm)
#pragma unroll
    for (int sn = 0; sn < 2; ++sn) {
      int col = n0 + wn * 32 + sn * 16 + l16;
#pragma unroll
      for (int r = 0; r < 8; ++r) {
        int row = q0 + wm * 32 + sm * 16 + g * 8 + r;
        float y = acc[sm][sn][r] * inv_sqrt_dh
                + adjm[((size_t)h * CN + row) * CN + col];
        S[((size_t)bh * CN + row) * CN + col] = y;
      }
    }
}

// ---------------------------------------------------------------------------
// AV: O[q,d] = sum_k P[q,k] * f_v[k,d]; per (b,h): [512x512] @ [512x128]
// P is f32 (fr_graph in d_out), converted to f16 during staging.
// V tile staged row-major [kk][n]; B fragments gathered column-wise.
// ---------------------------------------------------------------------------
__global__ __launch_bounds__(256) void av_gemm(
    const float* __restrict__ P, const _Float16* __restrict__ f_v,
    _Float16* __restrict__ x_pre)
{
  __shared__ _Float16 As[128 * LDSW];
  __shared__ _Float16 Vs[64 * LDSW];   // [kk][n] : 64 k-rows x 64 dh-cols
  const int tid = threadIdx.x;
  const int q0  = blockIdx.y * 128;
  const int n0  = blockIdx.x * 64;     // dh tile
  const int bh  = blockIdx.z;
  const int b   = bh / CH, h = bh % CH;
  const float*    Pbase = P   + (size_t)bh * CN * CN;
  const _Float16* Vbase = f_v + (size_t)b * CN * CD + h * CDH;

  const int wave = tid >> 5, lane = tid & 31;
  const int wm = wave & 3, wn = wave >> 2;
  const int g  = lane >> 4, l16 = lane & 15;

  v8h a_reg[4], b_reg[2];
  auto loadA = [&](int k0) {
#pragma unroll
    for (int p = 0; p < 4; ++p) {
      int idx = tid + p * 256;
      int m = idx >> 3, kk = (idx & 7) * 8;
      const float* src = Pbase + (size_t)(q0 + m) * CN + k0 + kk;
      a_reg[p] = cvt8(*(const v4f*)src, *(const v4f*)(src + 4));
    }
  };
  auto loadB = [&](int k0) {
#pragma unroll
    for (int p = 0; p < 2; ++p) {
      int idx = tid + p * 256;
      int kk = idx >> 3, n = (idx & 7) * 8;   // V row kk, contiguous n
      b_reg[p] = *(const v8h*)(Vbase + (size_t)(k0 + kk) * CD + n0 + n);
    }
  };
  auto storeLDS = [&]() {
#pragma unroll
    for (int p = 0; p < 4; ++p) {
      int idx = tid + p * 256;
      int m = idx >> 3, kk = (idx & 7) * 8;
      *(v8h*)(&As[m * LDSW + kk]) = a_reg[p];
    }
#pragma unroll
    for (int p = 0; p < 2; ++p) {
      int idx = tid + p * 256;
      int kk = idx >> 3, n = (idx & 7) * 8;
      *(v8h*)(&Vs[kk * LDSW + n]) = b_reg[p];
    }
  };

  v8f acc[2][2] = {};
  loadA(0); loadB(0);
  const int nkt = CN / KT;   // 8
  for (int kt = 0; kt < nkt; ++kt) {
    storeLDS();
    __syncthreads();
    if (kt + 1 < nkt) { loadA((kt + 1) * KT); loadB((kt + 1) * KT); }
#pragma unroll
    for (int ks = 0; ks < 2; ++ks) {
      const _Float16* ab = &As[ks * 32];
      const _Float16* vb = &Vs[ks * 32 * LDSW];
      v16h a0 = frag_a_row(ab + (wm * 32 + l16) * LDSW, g);
      v16h a1 = frag_a_row(ab + (wm * 32 + 16 + l16) * LDSW, g);
      v16h b0 = frag_b_col(vb, wn * 32 + l16, g);
      v16h b1 = frag_b_col(vb, wn * 32 + 16 + l16, g);
      acc[0][0] = WMMA_F32_F16(a0, b0, acc[0][0]);
      acc[0][1] = WMMA_F32_F16(a0, b1, acc[0][1]);
      acc[1][0] = WMMA_F32_F16(a1, b0, acc[1][0]);
      acc[1][1] = WMMA_F32_F16(a1, b1, acc[1][1]);
    }
    __syncthreads();
  }

#pragma unroll
  for (int sm = 0; sm < 2; ++sm)
#pragma unroll
    for (int sn = 0; sn < 2; ++sn) {
      int col = n0 + wn * 32 + sn * 16 + l16;     // dh index
#pragma unroll
      for (int r = 0; r < 8; ++r) {
        int row = q0 + wm * 32 + sm * 16 + g * 8 + r;
        x_pre[((size_t)b * CN + row) * CD + h * CDH + col] = (_Float16)acc[sm][sn][r];
      }
    }
}

// ---------------------------------------------------------------------------
// adj_mask[h,q,k] = (1 - softmax_k(diag0(col_head[h,q] . col_tail[h,k]))) * -1e4
// ---------------------------------------------------------------------------
__global__ __launch_bounds__(256) void adj_mask_kernel(
    const float* __restrict__ col_head, const float* __restrict__ col_tail,
    float* __restrict__ adjm)
{
  __shared__ float red[256];
  const int tid = threadIdx.x;
  const int h = blockIdx.x >> 9;
  const int q = blockIdx.x & 511;

  const float* ch = col_head + ((size_t)h * CN + q) * CDCOL;
  float chv[CDCOL];
#pragma unroll
  for (int i = 0; i < CDCOL; ++i) chv[i] = ch[i];

  float lg[2];
#pragma unroll
  for (int j = 0; j < 2; ++j) {
    int k = tid + j * 256;
    const float* ct = col_tail + ((size_t)h * CN + k) * CDCOL;
    float s = 0.0f;
#pragma unroll
    for (int i = 0; i < CDCOL; ++i) s += chv[i] * ct[i];
    lg[j] = (k == q) ? 0.0f : s;   // nsi: zero diagonal before softmax
  }

  red[tid] = fmaxf(lg[0], lg[1]); __syncthreads();
  for (int s = 128; s > 0; s >>= 1) { if (tid < s) red[tid] = fmaxf(red[tid], red[tid + s]); __syncthreads(); }
  float mx = red[0]; __syncthreads();

  float e0 = expf(lg[0] - mx), e1 = expf(lg[1] - mx);
  red[tid] = e0 + e1; __syncthreads();
  for (int s = 128; s > 0; s >>= 1) { if (tid < s) red[tid] += red[tid + s]; __syncthreads(); }
  float inv = 1.0f / red[0];

  float* out = adjm + ((size_t)h * CN + q) * CN;
  out[tid]       = (1.0f - e0 * inv) * (-10000.0f);
  out[tid + 256] = (1.0f - e1 * inv) * (-10000.0f);
}

// ---------------------------------------------------------------------------
// In-place row softmax over 512 columns (one block per row).
// ---------------------------------------------------------------------------
__global__ __launch_bounds__(256) void softmax_rows(float* __restrict__ P)
{
  __shared__ float red[256];
  const int tid = threadIdx.x;
  float* row = P + (size_t)blockIdx.x * CN;
  float x0 = row[tid], x1 = row[tid + 256];

  red[tid] = fmaxf(x0, x1); __syncthreads();
  for (int s = 128; s > 0; s >>= 1) { if (tid < s) red[tid] = fmaxf(red[tid], red[tid + s]); __syncthreads(); }
  float mx = red[0]; __syncthreads();

  float e0 = expf(x0 - mx), e1 = expf(x1 - mx);
  red[tid] = e0 + e1; __syncthreads();
  for (int s = 128; s > 0; s >>= 1) { if (tid < s) red[tid] += red[tid + s]; __syncthreads(); }
  float inv = 1.0f / red[0];

  row[tid]       = e0 * inv;
  row[tid + 256] = e1 * inv;
}

// ---------------------------------------------------------------------------
extern "C" void kernel_launch(void* const* d_in, const int* in_sizes, int n_in,
                              void* d_out, int out_size, void* d_ws, size_t ws_size,
                              hipStream_t stream)
{
  const float* x_q      = (const float*)d_in[0];
  const float* x_kv     = (const float*)d_in[1];
  const float* W_head_w = (const float*)d_in[2];
  const float* W_head_b = (const float*)d_in[3];
  const float* W_v_w    = (const float*)d_in[4];
  const float* W_v_b    = (const float*)d_in[5];
  const float* W_out_w  = (const float*)d_in[6];
  const float* W_out_b  = (const float*)d_in[7];
  const float* rel_emb  = (const float*)d_in[8];   // [H,DH] flat == per-column scale of length D
  const float* col_head = (const float*)d_in[9];
  const float* col_tail = (const float*)d_in[10];

  // workspace layout
  char* ws = (char*)d_ws;
  const size_t f16_mat = (size_t)CB * CN * CD * sizeof(_Float16);   // 33,554,432 B
  _Float16* f_head = (_Float16*)(ws);
  _Float16* f_tail = (_Float16*)(ws + f16_mat);
  _Float16* f_v    = (_Float16*)(ws + 2 * f16_mat);
  _Float16* x_pre  = (_Float16*)(ws + 3 * f16_mat);
  float*    adjm   = (float*)   (ws + 4 * f16_mat);                 // 8,388,608 B

  float* out_x = (float*)d_out;                       // [B,NQ,D]      = 16,777,216 f32
  float* fr    = out_x + (size_t)CB * CN * CD;        // [B*H,NQ,NKV]  = 67,108,864 f32

  const int M = CB * CN;   // 16384
  dim3 blk(256);
  dim3 gproj(CD / 64, M / 128);                       // 16 x 128

  // 1-3: projections (f16 outputs in ws)
  proj_gemm<0, 1><<<gproj, blk, 0, stream>>>(x_q,  W_head_w, W_head_b, rel_emb,
                                             f_head, M, CD, CD);
  proj_gemm<0, 1><<<gproj, blk, 0, stream>>>(x_kv, W_head_w, W_head_b, rel_emb,
                                             f_tail, M, CD, CD);
  proj_gemm<0, 1><<<gproj, blk, 0, stream>>>(x_kv, W_v_w, W_v_b, nullptr,
                                             f_v, M, CD, CD);

  // 4: learned topology mask (tiny, VALU)
  adj_mask_kernel<<<CH * CN, blk, 0, stream>>>(col_head, col_tail, adjm);

  // 5: scores straight into the fr_graph output region
  score_gemm<<<dim3(CN / 64, CN / 128, CB * CH), blk, 0, stream>>>(f_head, f_tail, adjm, fr);

  // 6: in-place softmax -> fr_graph (final output)
  softmax_rows<<<CB * CH * CN, blk, 0, stream>>>(fr);

  // 7: O = P @ V  (f16 into x_pre)
  av_gemm<<<dim3(CDH / 64, CN / 128, CB * CH), blk, 0, stream>>>(fr, f_v, x_pre);

  // 8: output projection (f32 into d_out)
  proj_gemm<1, 0><<<gproj, blk, 0, stream>>>(x_pre, W_out_w, W_out_b, nullptr,
                                             out_x, M, CD, CD);
}
